// Translayer_74775380623734
// MI455X (gfx1250) — matrix operand order repaired and checked by reading, not compiled, not run on hardware
//
#include <hip/hip_runtime.h>
#include <hip/hip_bf16.h>

// ---------------------------------------------------------------------------
// MI455X (gfx1250, wave32) transformer layer.
// All GEMMs run on v_wmma_f32_16x16x32_bf16 (f32 accumulate).
// Operand tensors are stored K-contiguous (i.e. "transposed" activations) so
// LDS staging is plain 16B copies and WMMA fragments are per-lane b128 LDS
// loads matching the ISA VGPR layouts (CDNA5 ISA §7.12.2).
// ---------------------------------------------------------------------------

typedef __attribute__((ext_vector_type(16))) __bf16 bf16x16;
typedef __attribute__((ext_vector_type(8)))  __bf16 bf16x8;
typedef __attribute__((ext_vector_type(8)))  float  f32x8;
typedef __attribute__((ext_vector_type(4)))  float  f32x4;

#define NB 8      // batch
#define DD 256    // channels / head dim
#define HH 8      // heads
#define PP 1024   // positions

__device__ inline f32x8 zero8() {
  f32x8 z;
  #pragma unroll
  for (int i = 0; i < 8; ++i) z[i] = 0.f;
  return z;
}

__device__ inline bf16x8 cvt8(f32x8 v) {
  bf16x8 r;
  #pragma unroll
  for (int i = 0; i < 8; ++i) r[i] = (__bf16)v[i];
  return r;
}

__device__ inline f32x8 wmma_bf16(bf16x16 a, bf16x16 b, f32x8 c) {
  return __builtin_amdgcn_wmma_f32_16x16x32_bf16(false, a, false, b,
                                                 (short)0, c, false, false);
}

// A fragment, 16x32 bf16 tile, LDS row-major (K contiguous), per-lane layout:
// lane = M (mod 16); half h: VGPR0-3 = K h*8..h*8+7, VGPR4-7 = K 16+h*8..+7.
__device__ inline bf16x16 frag_a(const __bf16* base, int stride, int lane) {
  int r = lane & 15, h = (lane >> 4) & 1;
  bf16x8 p0 = *(const bf16x8*)(base + r * stride + h * 8);
  bf16x8 p1 = *(const bf16x8*)(base + r * stride + h * 8 + 16);
  return __builtin_shufflevector(p0, p1, 0,1,2,3,4,5,6,7,8,9,10,11,12,13,14,15);
}

// B fragment, 32x16 bf16 tile, LDS stored [N][K] (K contiguous):
// lane = N (mod 16); half h: VGPR0-3 = K h*8..h*8+7, VGPR4-7 = K 16+h*8..+7.
__device__ inline bf16x16 frag_b(const __bf16* base, int stride, int lane) {
  int c = lane & 15, h = (lane >> 4) & 1;
  bf16x8 p0 = *(const bf16x8*)(base + c * stride + h * 8);
  bf16x8 p1 = *(const bf16x8*)(base + c * stride + h * 8 + 16);
  return __builtin_shufflevector(p0, p1, 0,1,2,3,4,5,6,7,8,9,10,11,12,13,14,15);
}

// ---------------------------------------------------------------------------
// f32 -> bf16 weight conversion
__global__ __launch_bounds__(256) void cvt_kernel(const float* __restrict__ s,
                                                  __bf16* __restrict__ d, int n) {
  int i = blockIdx.x * 256 + threadIdx.x;
  if (i < n) d[i] = (__bf16)s[i];
}

// ---------------------------------------------------------------------------
// LayerNorm over channel dim; in: [n,256,1024] f32, out: [n,1024,256] bf16.
__global__ __launch_bounds__(256) void ln_kernel(const float* __restrict__ x,
                                                 const float* __restrict__ g,
                                                 const float* __restrict__ b,
                                                 __bf16* __restrict__ outT) {
  int n = blockIdx.x >> 2;
  int p = (blockIdx.x & 3) * 256 + threadIdx.x;
  const float* xp = x + (size_t)n * DD * PP + p;
  float s = 0.f, s2 = 0.f;
  for (int d = 0; d < DD; ++d) { float v = xp[(size_t)d * PP]; s += v; s2 += v * v; }
  float mu = s * (1.f / DD);
  float var = s2 * (1.f / DD) - mu * mu;
  float rin = rsqrtf(var + 1e-5f);
  __bf16* op = outT + ((size_t)n * PP + p) * DD;
  for (int d0 = 0; d0 < DD; d0 += 8) {
    bf16x8 t;
    #pragma unroll
    for (int j = 0; j < 8; ++j) {
      int d = d0 + j;
      t[j] = (__bf16)((xp[(size_t)d * PP] - mu) * rin * g[d] + b[d]);
    }
    *(bf16x8*)(op + d0) = t;
  }
}

// ---------------------------------------------------------------------------
// Generic WMMA GEMM: out[o,p] = sum_k A[o,k] * Bt[n][p][k] + bias[o]
// Block = 128x128 tile, 256 threads (8 waves), each wave 4x2 16x16 tiles.
// Epilogue modes:
//  0: Q/K  -> bf16 [n, h, p, 256]  (o = h*256+d)
//  1: V    -> bf16 [n, h, 256, p]
//  2: Wo   -> res f32 [n,256,p] = acc + bias + x
//  3: MLP1 -> relu, bf16 [n, p, 256]
//  4: MLP2 -> outF f32 [n,256,p] = acc + bias + res
#define GST 40  // LDS K-stride (pad; keeps 16B alignment, spreads banks)
__global__ __launch_bounds__(256) void gemm_kernel(
    const __bf16* __restrict__ A, const __bf16* __restrict__ Bt,
    const float* __restrict__ bias, const float* __restrict__ aux,
    __bf16* __restrict__ outB, float* __restrict__ outF,
    int M, int K, int mode) {
  __shared__ __align__(16) __bf16 Al[128 * GST];
  __shared__ __align__(16) __bf16 Bl[128 * GST];

  int p0 = blockIdx.x * 128, m0 = blockIdx.y * 128, n = blockIdx.z;
  int tid = threadIdx.x, wv = tid >> 5, lane = tid & 31;
  int mgrp = wv & 1, ngrp = wv >> 1;
  int lhi = lane >> 4, llo = lane & 15;

  f32x8 acc[4][2];
  #pragma unroll
  for (int i = 0; i < 4; ++i)
    #pragma unroll
    for (int j = 0; j < 2; ++j) acc[i][j] = zero8();

  for (int k0 = 0; k0 < K; k0 += 32) {
    #pragma unroll
    for (int i = 0; i < 2; ++i) {
      int c = tid + i * 256;
      int row = c >> 2, ko = (c & 3) * 8;
      *(bf16x8*)&Al[row * GST + ko] =
          *(const bf16x8*)&A[(size_t)(m0 + row) * K + k0 + ko];
      *(bf16x8*)&Bl[row * GST + ko] =
          *(const bf16x8*)&Bt[((size_t)n * PP + p0 + row) * K + k0 + ko];
    }
    __syncthreads();
    bf16x16 af[4], bf[2];
    #pragma unroll
    for (int mi = 0; mi < 4; ++mi)
      af[mi] = frag_a(Al + (mgrp * 64 + mi * 16) * GST, GST, lane);
    #pragma unroll
    for (int ni = 0; ni < 2; ++ni)
      bf[ni] = frag_b(Bl + (ngrp * 32 + ni * 16) * GST, GST, lane);
    #pragma unroll
    for (int mi = 0; mi < 4; ++mi)
      #pragma unroll
      for (int ni = 0; ni < 2; ++ni)
        acc[mi][ni] = wmma_bf16(af[mi], bf[ni], acc[mi][ni]);
    __syncthreads();
  }

  // epilogue
  #pragma unroll
  for (int mi = 0; mi < 4; ++mi) {
    #pragma unroll
    for (int ni = 0; ni < 2; ++ni) {
      int obase = m0 + (mgrp * 4 + mi) * 16 + lhi * 8;
      int p = p0 + (ngrp * 2 + ni) * 16 + llo;
      f32x8 r = acc[mi][ni];
      #pragma unroll
      for (int v = 0; v < 8; ++v) r[v] += bias[obase + v];
      if (mode == 0) {
        int h = obase >> 8, d = obase & 255;
        __bf16* dst = outB + (((size_t)(n * HH + h)) * PP + p) * DD + d;
        *(bf16x8*)dst = cvt8(r);
      } else if (mode == 1) {
        #pragma unroll
        for (int v = 0; v < 8; ++v) {
          int o = obase + v;
          outB[(((size_t)(n * HH + (o >> 8))) * DD + (o & 255)) * PP + p] =
              (__bf16)r[v];
        }
      } else if (mode == 2) {
        #pragma unroll
        for (int v = 0; v < 8; ++v) {
          size_t idx = ((size_t)n * DD + obase + v) * PP + p;
          outF[idx] = r[v] + aux[idx];
        }
      } else if (mode == 3) {
        #pragma unroll
        for (int v = 0; v < 8; ++v) r[v] = fmaxf(r[v], 0.f);
        __bf16* dst = outB + ((size_t)n * PP + p) * DD + obase;
        *(bf16x8*)dst = cvt8(r);
      } else {  // mode 4
        #pragma unroll
        for (int v = 0; v < 8; ++v) {
          size_t idx = ((size_t)n * DD + obase + v) * PP + p;
          outF[idx] = r[v] + aux[idx];
        }
      }
    }
  }
}

// ---------------------------------------------------------------------------
// Flash-style attention. Grid = (16 qblocks, 8 heads, 8 batch), 256 threads.
// qt/kt: [n,h,1024,256] bf16 (d contiguous); vb: [n,h,256,1024] bf16;
// sat out: [n,1024,2048] bf16 (c = h*256+d contiguous).
// LDS: V-tile (36864B) overlaps S (18432B); + W 9216 + K 5120 + Q 5120 + 2048.
__global__ __launch_bounds__(256) void attn_kernel(
    const __bf16* __restrict__ qt, const __bf16* __restrict__ kt,
    const __bf16* __restrict__ vb, const float* __restrict__ pos,
    __bf16* __restrict__ sat) {
  __shared__ __align__(16) unsigned char smem[58368];
  __bf16* Vl = (__bf16*)smem;                 // [256][72]
  float*  Sl = (float*)smem;                  // [64][72] (overlaps Vl)
  __bf16* Wl = (__bf16*)(smem + 36864);       // [64][72]
  __bf16* Kl = (__bf16*)(smem + 46080);       // [64][40]
  __bf16* Ql = (__bf16*)(smem + 51200);       // [64][40]
  float* MR  = (float*)(smem + 56320);
  float* LR  = MR + 64;
  float* AL  = LR + 64;
  float* MN  = AL + 64;
  float* SCR = MN + 64;                       // [4][64]

  int q0 = blockIdx.x * 64, h = blockIdx.y, n = blockIdx.z;
  int tid = threadIdx.x, wv = tid >> 5, lane = tid & 31;
  int lhi = lane >> 4, llo = lane & 15;
  int tm = wv >> 1, tn0 = (wv & 1) * 2;       // S-tile assignment
  int sq = tid & 63, seg = tid >> 6;          // softmax assignment

  if (tid < 64) { MR[tid] = -__builtin_inff(); LR[tid] = 0.f; }

  f32x8 oacc[2][4];
  #pragma unroll
  for (int mt = 0; mt < 2; ++mt)
    #pragma unroll
    for (int tn = 0; tn < 4; ++tn) oacc[mt][tn] = zero8();

  const size_t nh = (size_t)(n * HH + h);
  const float* ph = pos + (size_t)h * PP * PP;

  for (int kb = 0; kb < 16; ++kb) {
    int k0 = kb * 64;
    // ---- phase A: S[64k x 64q] = K^T Q, d chunked by 32 through LDS ----
    f32x8 sacc[2];
    sacc[0] = zero8(); sacc[1] = zero8();
    for (int dc = 0; dc < 8; ++dc) {
      int d0 = dc * 32;
      int row = tid >> 2, ko = (tid & 3) * 8;
      *(bf16x8*)&Kl[row * 40 + ko] =
          *(const bf16x8*)&kt[(nh * PP + k0 + row) * DD + d0 + ko];
      *(bf16x8*)&Ql[row * 40 + ko] =
          *(const bf16x8*)&qt[(nh * PP + q0 + row) * DD + d0 + ko];
      __syncthreads();
      bf16x16 a  = frag_a(Kl + tm * 16 * 40, 40, lane);
      bf16x16 b0 = frag_b(Ql + tn0 * 16 * 40, 40, lane);
      bf16x16 b1 = frag_b(Ql + (tn0 + 1) * 16 * 40, 40, lane);
      sacc[0] = wmma_bf16(a, b0, sacc[0]);
      sacc[1] = wmma_bf16(a, b1, sacc[1]);
      __syncthreads();
    }
    // write S tiles to LDS as [q][k]
    #pragma unroll
    for (int i = 0; i < 2; ++i) {
      int q = (tn0 + i) * 16 + llo;
      int kr = tm * 16 + lhi * 8;
      float* dst = &Sl[q * 72 + kr];
      *(f32x4*)dst       = __builtin_shufflevector(sacc[i], sacc[i], 0, 1, 2, 3);
      *(f32x4*)(dst + 4) = __builtin_shufflevector(sacc[i], sacc[i], 4, 5, 6, 7);
    }
    __syncthreads();

    // ---- online softmax over this key block ----
    float mx = -__builtin_inff();
    for (int j = 0; j < 16; ++j) {
      int kk = seg * 16 + j;
      float s = Sl[sq * 72 + kk] * 0.0625f +
                ph[(size_t)(k0 + kk) * PP + q0 + sq];
      Sl[sq * 72 + kk] = s;
      mx = fmaxf(mx, s);
    }
    SCR[seg * 64 + sq] = mx;
    __syncthreads();
    if (tid < 64) {
      float mb = fmaxf(fmaxf(SCR[tid], SCR[64 + tid]),
                       fmaxf(SCR[128 + tid], SCR[192 + tid]));
      float mo = MR[tid];
      float mn = fmaxf(mo, mb);
      MN[tid] = mn;
      AL[tid] = __expf(mo - mn);   // 0 on first block
      MR[tid] = mn;
    }
    __syncthreads();
    float mn = MN[sq], ls = 0.f;
    for (int j = 0; j < 16; ++j) {
      int kk = seg * 16 + j;
      float w = __expf(Sl[sq * 72 + kk] - mn);
      Wl[sq * 72 + kk] = (__bf16)w;   // [q][k], k contiguous = B layout
      ls += w;
    }
    SCR[seg * 64 + sq] = ls;
    __syncthreads();
    if (tid < 64)
      LR[tid] = LR[tid] * AL[tid] +
                SCR[tid] + SCR[64 + tid] + SCR[128 + tid] + SCR[192 + tid];
    __syncthreads();

    // ---- phase B: stage V tile (overlaps S), rescale, O += V * W ----
    #pragma unroll
    for (int i = 0; i < 8; ++i) {
      int c = tid + i * 256;
      int row = c >> 3, ko = (c & 7) * 8;
      *(bf16x8*)&Vl[row * 72 + ko] =
          *(const bf16x8*)&vb[(nh * DD + row) * PP + k0 + ko];
    }
    __syncthreads();
    #pragma unroll
    for (int tn = 0; tn < 4; ++tn) {
      float a = AL[tn * 16 + llo];
      #pragma unroll
      for (int mt = 0; mt < 2; ++mt)
        #pragma unroll
        for (int e = 0; e < 8; ++e) oacc[mt][tn][e] *= a;
    }
    #pragma unroll
    for (int ks = 0; ks < 2; ++ks) {
      bf16x16 bfr[4];
      #pragma unroll
      for (int tn = 0; tn < 4; ++tn)
        bfr[tn] = frag_b(Wl + tn * 16 * 72 + ks * 32, 72, lane);
      #pragma unroll
      for (int mt = 0; mt < 2; ++mt) {
        bf16x16 a = frag_a(Vl + (wv * 32 + mt * 16) * 72 + ks * 32, 72, lane);
        #pragma unroll
        for (int tn = 0; tn < 4; ++tn)
          oacc[mt][tn] = wmma_bf16(a, bfr[tn], oacc[mt][tn]);
      }
    }
    __syncthreads();
  }

  // ---- epilogue: O /= l, write sat [n, p, h*256+d] ----
  #pragma unroll
  for (int tn = 0; tn < 4; ++tn) {
    float inv = 1.f / LR[tn * 16 + llo];
    int q = q0 + tn * 16 + llo;
    #pragma unroll
    for (int mt = 0; mt < 2; ++mt) {
      int d = wv * 32 + mt * 16 + lhi * 8;
      f32x8 r = oacc[mt][tn];
      #pragma unroll
      for (int e = 0; e < 8; ++e) r[e] *= inv;
      __bf16* dst = sat + ((size_t)n * PP + q) * (HH * DD) + h * DD + d;
      *(bf16x8*)dst = cvt8(r);
    }
  }
}

// ---------------------------------------------------------------------------
extern "C" void kernel_launch(void* const* d_in, const int* in_sizes, int n_in,
                              void* d_out, int out_size, void* d_ws, size_t ws_size,
                              hipStream_t stream) {
  const float* x    = (const float*)d_in[0];
  const float* pos  = (const float*)d_in[1];
  const float* Wq   = (const float*)d_in[2];
  const float* bq   = (const float*)d_in[3];
  const float* Wk   = (const float*)d_in[4];
  const float* bk   = (const float*)d_in[5];
  const float* Wv   = (const float*)d_in[6];
  const float* bv   = (const float*)d_in[7];
  const float* Wo   = (const float*)d_in[8];
  const float* bo   = (const float*)d_in[9];
  const float* g1   = (const float*)d_in[10];
  const float* b1n  = (const float*)d_in[11];
  const float* g2   = (const float*)d_in[12];
  const float* b2n  = (const float*)d_in[13];
  const float* W1   = (const float*)d_in[14];
  const float* b1   = (const float*)d_in[15];
  const float* W2   = (const float*)d_in[16];
  const float* b2   = (const float*)d_in[17];
  float* out = (float*)d_out;

  char* ws = (char*)d_ws;
  size_t off = 0;
  auto alloc = [&](size_t bytes) { char* p = ws + off; off += (bytes + 255) & ~(size_t)255; return p; };
  __bf16* wq_bf = (__bf16*)alloc(2048 * 256 * 2);
  __bf16* wk_bf = (__bf16*)alloc(2048 * 256 * 2);
  __bf16* wv_bf = (__bf16*)alloc(2048 * 256 * 2);
  __bf16* wo_bf = (__bf16*)alloc(256 * 2048 * 2);
  __bf16* w1_bf = (__bf16*)alloc(256 * 256 * 2);
  __bf16* w2_bf = (__bf16*)alloc(256 * 256 * 2);
  __bf16* x1t   = (__bf16*)alloc((size_t)NB * PP * DD * 2);
  __bf16* qT    = (__bf16*)alloc((size_t)NB * HH * PP * DD * 2);
  __bf16* kT    = (__bf16*)alloc((size_t)NB * HH * PP * DD * 2);
  __bf16* vB    = (__bf16*)alloc((size_t)NB * HH * DD * PP * 2);
  __bf16* sat   = (__bf16*)alloc((size_t)NB * PP * HH * DD * 2);
  float*  res   = (float*)alloc((size_t)NB * DD * PP * 4);
  __bf16* x2t   = (__bf16*)alloc((size_t)NB * PP * DD * 2);
  __bf16* m1t   = (__bf16*)alloc((size_t)NB * PP * DD * 2);

  // weight conversion
  cvt_kernel<<<2048, 256, 0, stream>>>(Wq, wq_bf, 2048 * 256);
  cvt_kernel<<<2048, 256, 0, stream>>>(Wk, wk_bf, 2048 * 256);
  cvt_kernel<<<2048, 256, 0, stream>>>(Wv, wv_bf, 2048 * 256);
  cvt_kernel<<<2048, 256, 0, stream>>>(Wo, wo_bf, 256 * 2048);
  cvt_kernel<<<256, 256, 0, stream>>>(W1, w1_bf, 256 * 256);
  cvt_kernel<<<256, 256, 0, stream>>>(W2, w2_bf, 256 * 256);

  // LN1
  ln_kernel<<<32, 256, 0, stream>>>(x, g1, b1n, x1t);

  // QKV projections
  gemm_kernel<<<dim3(8, 16, 8), 256, 0, stream>>>(wq_bf, x1t, bq, nullptr, qT, nullptr, 2048, 256, 0);
  gemm_kernel<<<dim3(8, 16, 8), 256, 0, stream>>>(wk_bf, x1t, bk, nullptr, kT, nullptr, 2048, 256, 0);
  gemm_kernel<<<dim3(8, 16, 8), 256, 0, stream>>>(wv_bf, x1t, bv, nullptr, vB, nullptr, 2048, 256, 1);

  // attention
  attn_kernel<<<dim3(16, 8, 8), 256, 0, stream>>>(qT, kT, vB, pos, sat);

  // output projection + residual
  gemm_kernel<<<dim3(8, 2, 8), 256, 0, stream>>>(wo_bf, sat, bo, x, nullptr, res, 256, 2048, 2);

  // LN2
  ln_kernel<<<32, 256, 0, stream>>>(res, g2, b2n, x2t);

  // MLP
  gemm_kernel<<<dim3(8, 2, 8), 256, 0, stream>>>(w1_bf, x2t, b1, nullptr, m1t, nullptr, 256, 256, 3);
  gemm_kernel<<<dim3(8, 2, 8), 256, 0, stream>>>(w2_bf, m1t, b2, res, nullptr, out, 256, 256, 4);
}
